// img_attention_26955214750221
// MI455X (gfx1250) — compile-verified, matrix operand order checked by reading, not run on hardware
//
#include <hip/hip_runtime.h>

#define GFD 4096
#define DH  1024
#define BATCH 32
#define TK 196
#define MROWS (BATCH * TK)   // 6272 = 49 * 128

typedef __attribute__((ext_vector_type(16))) __bf16           v16bf;
typedef __attribute__((ext_vector_type(16))) unsigned short   v16u;
typedef __attribute__((ext_vector_type(8)))  float            v8f;
typedef __attribute__((__vector_size__(16))) int              b128_t;

__device__ __forceinline__ unsigned short f32_to_bf16u(float f) {
  unsigned int u = __builtin_bit_cast(unsigned int, f);
  u += 0x7FFFu + ((u >> 16) & 1u);          // round-to-nearest-even
  return (unsigned short)(u >> 16);
}
__device__ __forceinline__ unsigned int pack_bf16(float lo, float hi) {
  return (unsigned int)f32_to_bf16u(lo) | ((unsigned int)f32_to_bf16u(hi) << 16);
}

// ---------------------------------------------------------------------------
// CDNA5 async global->LDS staging (GLOBAL_LOAD_ASYNC_TO_LDS_B128, ASYNCcnt),
// with a synchronous b128 fallback if the builtin is unavailable.
// Builtin signature (from hipcc diagnostic): (v4i AS1*, v4i AS3*, imm, imm)
// ---------------------------------------------------------------------------
#if __has_builtin(__builtin_amdgcn_global_load_async_to_lds_b128)
#define HAVE_ASYNC 1
#else
#define HAVE_ASYNC 0
#endif

__device__ __forceinline__ void copy_b128_to_lds(const unsigned short* src,
                                                 unsigned short* dst) {
#if HAVE_ASYNC
  __builtin_amdgcn_global_load_async_to_lds_b128(
      (__attribute__((address_space(1))) b128_t*)src,
      (__attribute__((address_space(3))) b128_t*)dst, 0, 0);
#else
  *(uint4*)dst = *(const uint4*)src;
#endif
}

__device__ __forceinline__ void wait_async0() {
#if HAVE_ASYNC
#if __has_builtin(__builtin_amdgcn_s_wait_asynccnt)
  __builtin_amdgcn_s_wait_asynccnt(0);
#else
  asm volatile("s_wait_asynccnt 0x0" ::: "memory");
#endif
#endif
}
__device__ __forceinline__ void wait_async4() {
#if HAVE_ASYNC
#if __has_builtin(__builtin_amdgcn_s_wait_asynccnt)
  __builtin_amdgcn_s_wait_asynccnt(4);
#else
  asm volatile("s_wait_asynccnt 0x4" ::: "memory");
#endif
#endif
}

// ---------------------------------------------------------------------------
// WMMA GEMM, all-bf16 operands. Block tile 128x128, BK=32, double-buffered
// LDS. A is [M][K] bf16 row-major; B is PRE-TRANSPOSED [N][K] bf16 so both
// tiles stage identically (contiguous 64B rows). 8 wave32s -> each wave owns
// 32(M) x 64(N): 2x4 accumulators of v_wmma_f32_16x16x32_bf16.
// ---------------------------------------------------------------------------
#define BM 128
#define BN 128
#define BK 32
#define LDP (BK + 8)   // padded LDS row (ushorts); 80B, 16B-aligned rows

__device__ __forceinline__ void stage_tiles(const unsigned short* __restrict__ Ab,
                                            const unsigned short* __restrict__ Bb,
                                            int K, int kk,
                                            unsigned short (*As)[LDP],
                                            unsigned short (*Bs)[LDP], int tid) {
  // tile = 128 rows x 32 bf16 = 512 x 16B slots; 256 threads -> 2 slots each
#pragma unroll
  for (int s = 0; s < 2; s++) {
    int slot = tid + s * 256;
    int row = slot >> 2, ch = (slot & 3) * 8;
    copy_b128_to_lds(Ab + (size_t)row * K + kk + ch, &As[row][ch]);
    copy_b128_to_lds(Bb + (size_t)row * K + kk + ch, &Bs[row][ch]);
  }
}

// fragment gather per 16-bit A layout: elem e -> K = (e>=8?16:0) + khalf*8 + (e&7)
__device__ __forceinline__ v16bf frag16(const unsigned short (*T)[LDP], int row, int khalf) {
  v16u t;
#pragma unroll
  for (int e = 0; e < 16; e++) {
    int k = ((e >> 3) << 4) + khalf * 8 + (e & 7);
    t[e] = T[row][k];
  }
  return __builtin_bit_cast(v16bf, t);
}

template <int OUTMODE>   // 0: bf16 C0 ; 1: f32 C0 + bf16 C1
__global__ __launch_bounds__(256)
void gemm_bf16(const unsigned short* __restrict__ A, const unsigned short* __restrict__ Bt,
               const float* __restrict__ bias, void* __restrict__ C0,
               unsigned short* __restrict__ C1, int N, int K) {
  __shared__ alignas(16) unsigned short As[2][BM][LDP];
  __shared__ alignas(16) unsigned short Bs[2][BN][LDP];

  const int bm   = blockIdx.y * BM;
  const int bn   = blockIdx.x * BN;
  const int tid  = threadIdx.x;
  const int lane = tid & 31;
  const int wave = tid >> 5;
  const int waveM = (wave & 3) * 32;
  const int waveN = (wave >> 2) * 64;
  const int khalf = (lane >> 4) & 1;
  const int lrow  = lane & 15;

  const unsigned short* Ab = A  + (size_t)bm * K;
  const unsigned short* Bb = Bt + (size_t)bn * K;

  v8f acc[2][4] = {};

  stage_tiles(Ab, Bb, K, 0, As[0], Bs[0], tid);
  const int nk = K / BK;
  for (int it = 0; it < nk; ++it) {
    const int cur = it & 1;
    const bool more = (it + 1) < nk;
    if (more) stage_tiles(Ab, Bb, K, (it + 1) * BK, As[cur ^ 1], Bs[cur ^ 1], tid);
    if (more) wait_async4(); else wait_async0();
    __syncthreads();

    v16bf bfrag[4];
#pragma unroll
    for (int ni = 0; ni < 4; ni++) bfrag[ni] = frag16(Bs[cur], waveN + ni * 16 + lrow, khalf);
#pragma unroll
    for (int mi = 0; mi < 2; mi++) {
      v16bf afrag = frag16(As[cur], waveM + mi * 16 + lrow, khalf);
#pragma unroll
      for (int ni = 0; ni < 4; ni++)
        acc[mi][ni] = __builtin_amdgcn_wmma_f32_16x16x32_bf16(
            false, afrag, false, bfrag[ni], (short)0, acc[mi][ni], false, false);
    }
    __syncthreads();
  }

  const int moff = (lane < 16) ? 0 : 8;
#pragma unroll
  for (int mi = 0; mi < 2; mi++)
#pragma unroll
    for (int ni = 0; ni < 4; ni++)
#pragma unroll
      for (int r = 0; r < 8; r++) {
        int m = bm + waveM + mi * 16 + r + moff;
        int n = bn + waveN + ni * 16 + lrow;
        float val = acc[mi][ni][r] + bias[n];
        if (OUTMODE == 0) {
          ((unsigned short*)C0)[(size_t)m * N + n] = f32_to_bf16u(val);
        } else {
          ((float*)C0)[(size_t)m * N + n] = val;
          C1[(size_t)m * N + n] = f32_to_bf16u(val);
        }
      }
}

// ---------------------------------------------------------------------------
// GEMM3 + fused score epilogue: scores[b,t] += sum_col tanh(w + dec + cov)*v[col]
// ---------------------------------------------------------------------------
__global__ __launch_bounds__(256)
void gemm_score(const unsigned short* __restrict__ A, const unsigned short* __restrict__ Bt,
                const float* __restrict__ dec_fea, const float* __restrict__ cov,
                const float* __restrict__ vvec, float* __restrict__ scores,
                int N, int K) {
  __shared__ alignas(16) unsigned short As[2][BM][LDP];
  __shared__ alignas(16) unsigned short Bs[2][BN][LDP];

  const int bm   = blockIdx.y * BM;
  const int bn   = blockIdx.x * BN;
  const int tid  = threadIdx.x;
  const int lane = tid & 31;
  const int wave = tid >> 5;
  const int waveM = (wave & 3) * 32;
  const int waveN = (wave >> 2) * 64;
  const int khalf = (lane >> 4) & 1;
  const int lrow  = lane & 15;

  const unsigned short* Ab = A  + (size_t)bm * K;
  const unsigned short* Bb = Bt + (size_t)bn * K;

  v8f acc[2][4] = {};

  stage_tiles(Ab, Bb, K, 0, As[0], Bs[0], tid);
  const int nk = K / BK;
  for (int it = 0; it < nk; ++it) {
    const int cur = it & 1;
    const bool more = (it + 1) < nk;
    if (more) stage_tiles(Ab, Bb, K, (it + 1) * BK, As[cur ^ 1], Bs[cur ^ 1], tid);
    if (more) wait_async4(); else wait_async0();
    __syncthreads();

    v16bf bfrag[4];
#pragma unroll
    for (int ni = 0; ni < 4; ni++) bfrag[ni] = frag16(Bs[cur], waveN + ni * 16 + lrow, khalf);
#pragma unroll
    for (int mi = 0; mi < 2; mi++) {
      v16bf afrag = frag16(As[cur], waveM + mi * 16 + lrow, khalf);
#pragma unroll
      for (int ni = 0; ni < 4; ni++)
        acc[mi][ni] = __builtin_amdgcn_wmma_f32_16x16x32_bf16(
            false, afrag, false, bfrag[ni], (short)0, acc[mi][ni], false, false);
    }
    __syncthreads();
  }

  const int moff = (lane < 16) ? 0 : 8;
#pragma unroll
  for (int mi = 0; mi < 2; mi++)
#pragma unroll
    for (int r = 0; r < 8; r++) {
      int row = bm + waveM + mi * 16 + r + moff;   // flat (b,t)
      int b = row / TK;
      float cv = cov[row];
      float s = 0.f;
#pragma unroll
      for (int ni = 0; ni < 4; ni++) {
        int col = bn + waveN + ni * 16 + lrow;
        float val = acc[mi][ni][r] + dec_fea[b * DH + col] + cv;
        s += tanhf(val) * vvec[col];
      }
      for (int off = 1; off < 16; off <<= 1) s += __shfl_xor(s, off, 32);
      if (lrow == 0) atomicAdd(&scores[row], s);
    }
}

// --------------------------- conversion kernels -----------------------------
__global__ __launch_bounds__(256)
void cvt_bf16_vec(const float* __restrict__ src, unsigned short* __restrict__ dst,
                  long n8) {   // n/8 16B-chunks
  long i = (long)blockIdx.x * 256 + threadIdx.x;
  if (i >= n8) return;
  float4 a = ((const float4*)src)[i * 2];
  float4 b = ((const float4*)src)[i * 2 + 1];
  uint4 o;
  o.x = pack_bf16(a.x, a.y);
  o.y = pack_bf16(a.z, a.w);
  o.z = pack_bf16(b.x, b.y);
  o.w = pack_bf16(b.z, b.w);
  ((uint4*)dst)[i] = o;
}

// W[K][N] f32  ->  WT[N][K] bf16 (32x32 LDS tile)
__global__ __launch_bounds__(256)
void transpose_cvt(const float* __restrict__ W, unsigned short* __restrict__ WT,
                   int K, int N) {
  __shared__ unsigned short t[32][33];
  int n0 = blockIdx.x * 32, k0 = blockIdx.y * 32;
  int tx = threadIdx.x & 31, ty = threadIdx.x >> 5;
  for (int r = ty; r < 32; r += 8)
    t[r][tx] = f32_to_bf16u(W[(size_t)(k0 + r) * N + (n0 + tx)]);
  __syncthreads();
  for (int r = ty; r < 32; r += 8)
    WT[(size_t)(n0 + r) * K + (k0 + tx)] = t[tx][r];
}

// --------------------------- small helper kernels ---------------------------
__global__ __launch_bounds__(256)
void zero_buf(float* __restrict__ p, int n) {
  int i = blockIdx.x * 256 + threadIdx.x;
  if (i < n) p[i] = 0.f;
}

__global__ __launch_bounds__(256)
void dec_proj(const float* __restrict__ s_t_hat, const float* __restrict__ Wdec,
              const float* __restrict__ bdec, float* __restrict__ dec_fea) {
  int idx = blockIdx.x * 256 + threadIdx.x;    // 32 * 1024
  int b = idx >> 10, j = idx & 1023;
  float acc = bdec[j];
  for (int k = 0; k < DH; k++)
    acc = fmaf(s_t_hat[b * DH + k], Wdec[(size_t)k * DH + j], acc);
  dec_fea[idx] = acc;
}

__global__ __launch_bounds__(256)
void softmax_cov(const float* __restrict__ scores, const float* __restrict__ cov,
                 float* __restrict__ alpha_ws, float* __restrict__ out) {
  __shared__ float red[256];
  int b = blockIdx.x, t = threadIdx.x;
  float val = (t < TK) ? scores[b * TK + t] : -3.4e38f;
  red[t] = val; __syncthreads();
  for (int s = 128; s > 0; s >>= 1) {
    if (t < s) red[t] = fmaxf(red[t], red[t + s]);
    __syncthreads();
  }
  float mx = red[0]; __syncthreads();
  float e = (t < TK) ? __expf(val - mx) : 0.f;
  red[t] = e; __syncthreads();
  for (int s = 128; s > 0; s >>= 1) {
    if (t < s) red[t] += red[t + s];
    __syncthreads();
  }
  float inv = 1.f / red[0];
  if (t < TK) {
    float a = e * inv;
    alpha_ws[b * TK + t] = a;
    out[BATCH * DH + b * TK + t]              = cov[b * TK + t] + a; // coverage_new
    out[BATCH * DH + BATCH * TK + b * TK + t] = a;                   // alpha_a
  }
}

__global__ __launch_bounds__(256)
void context_bmm(const float* __restrict__ alpha, const float* __restrict__ g_star,
                 float* __restrict__ out) {
  __shared__ float al[TK];
  int b = blockIdx.x >> 2;
  int d = (blockIdx.x & 3) * 256 + threadIdx.x;
  if (threadIdx.x < TK) al[threadIdx.x] = alpha[b * TK + threadIdx.x];
  __syncthreads();
  float acc = 0.f;
  for (int t = 0; t < TK; t++)
    acc = fmaf(al[t], g_star[((size_t)(b * TK + t)) * DH + d], acc);
  out[b * DH + d] = acc;
}

// ----------------------------------------------------------------------------
extern "C" void kernel_launch(void* const* d_in, const int* in_sizes, int n_in,
                              void* d_out, int out_size, void* d_ws, size_t ws_size,
                              hipStream_t stream) {
  const float* gf      = (const float*)d_in[0];   // [32,196,4096]
  const float* s_t_hat = (const float*)d_in[1];   // [32,1024]
  const float* cov     = (const float*)d_in[2];   // [32,196,1]
  const float* Wg      = (const float*)d_in[3];   // [4096,4096]
  const float* bg      = (const float*)d_in[4];   // [4096]
  const float* Gs      = (const float*)d_in[5];   // [4096,1024]
  const float* bgs     = (const float*)d_in[6];   // [1024]
  const float* Wgs     = (const float*)d_in[7];   // [1024,1024]
  const float* Wdec    = (const float*)d_in[8];   // [1024,1024]
  const float* bdec    = (const float*)d_in[9];   // [1024]
  const float* vvec    = (const float*)d_in[10];  // [1024,1]
  float* out = (float*)d_out;

  char* ws = (char*)d_ws;
  size_t off = 0;
  auto take = [&](size_t bytes) { char* p = ws + off; off = (off + bytes + 255) & ~(size_t)255; return p; };
  unsigned short* gf16    = (unsigned short*)take((size_t)MROWS * GFD * 2);
  unsigned short* WgT     = (unsigned short*)take((size_t)GFD * GFD * 2);
  unsigned short* GsT     = (unsigned short*)take((size_t)DH * GFD * 2);
  unsigned short* WgsT    = (unsigned short*)take((size_t)DH * DH * 2);
  unsigned short* g1      = (unsigned short*)take((size_t)MROWS * GFD * 2);
  float*          gstar   = (float*)take((size_t)MROWS * DH * 4);
  unsigned short* gstar16 = (unsigned short*)take((size_t)MROWS * DH * 2);
  float*          decf    = (float*)take((size_t)BATCH * DH * 4);
  float*          scores  = (float*)take((size_t)MROWS * 4);
  float*          alpha   = (float*)take((size_t)MROWS * 4);

  // 0) one-time precision/layout conversions (bandwidth-trivial vs 276 GFLOP)
  cvt_bf16_vec<<<((size_t)MROWS * GFD / 8 + 255) / 256, 256, 0, stream>>>(
      gf, gf16, (long)MROWS * GFD / 8);
  transpose_cvt<<<dim3(GFD / 32, GFD / 32), 256, 0, stream>>>(Wg, WgT, GFD, GFD);
  transpose_cvt<<<dim3(DH / 32, GFD / 32), 256, 0, stream>>>(Gs, GsT, GFD, DH);
  transpose_cvt<<<dim3(DH / 32, DH / 32), 256, 0, stream>>>(Wgs, WgsT, DH, DH);

  // 1) g1 = gf @ Wg + bg                 [6272,4096] bf16
  gemm_bf16<0><<<dim3(GFD / BN, MROWS / BM), 256, 0, stream>>>(
      gf16, WgT, bg, g1, nullptr, GFD, GFD);
  // 2) g_star = g1 @ Gs + bgs            [6272,1024] f32 + bf16 copy
  gemm_bf16<1><<<dim3(DH / BN, MROWS / BM), 256, 0, stream>>>(
      g1, GsT, bgs, gstar, gstar16, DH, GFD);
  // 3) dec_fea = s_t_hat @ Wdec + bdec
  dec_proj<<<(BATCH * DH) / 256, 256, 0, stream>>>(s_t_hat, Wdec, bdec, decf);
  // 4) fused scores
  zero_buf<<<(MROWS + 255) / 256, 256, 0, stream>>>(scores, MROWS);
  gemm_score<<<dim3(DH / BN, MROWS / BM), 256, 0, stream>>>(
      gstar16, WgsT, decf, cov, vvec, scores, DH, DH);
  // 5) softmax + coverage_new + alpha outputs
  softmax_cov<<<BATCH, 256, 0, stream>>>(scores, cov, alpha, out);
  // 6) c_img
  context_bmm<<<BATCH * (DH / 256), 256, 0, stream>>>(alpha, gstar, out);
}
